// LearnableSparseEncoder_17497696764488
// MI455X (gfx1250) — matrix operand order repaired and checked by reading, not compiled, not run on hardware
//
#include <hip/hip_runtime.h>
#include <hip/hip_bf16.h>
#include <math.h>

typedef __bf16 bf16_t;
typedef __attribute__((ext_vector_type(16))) __bf16 v16bf;
typedef __attribute__((ext_vector_type(8)))  float  v8f;
typedef __attribute__((ext_vector_type(4)))  unsigned int u32x4;

union FragBF {
    v16bf v;
    u32x4 u[2];
};

// ---------------------------------------------------------------------------
// f32 -> bf16 conversion (vectorized x4)
// ---------------------------------------------------------------------------
__global__ void __launch_bounds__(256)
cvt_f32_to_bf16(const float* __restrict__ in, bf16_t* __restrict__ out, int n4) {
    int i = blockIdx.x * 256 + threadIdx.x;
    if (i < n4) {
        const float4 f = ((const float4*)in)[i];
        union { bf16_t b[4]; unsigned long long q; } pk;
        pk.b[0] = (bf16_t)f.x; pk.b[1] = (bf16_t)f.y;
        pk.b[2] = (bf16_t)f.z; pk.b[3] = (bf16_t)f.w;
        *(unsigned long long*)(out + (size_t)i * 4) = pk.q;
    }
}

// ---------------------------------------------------------------------------
// bf16 WMMA GEMM with CDNA5 async global->LDS double buffering.
//   C[M,N] = alpha * (A[M,K] @ W[N,K]^T) + bias[N] + resid[M,N]
// Block tile 128x128, K-step 32, 8 waves (4x2), wave tile 32x64.
// Global->LDS uses GLOBAL_LOAD_ASYNC_TO_LDS_B128 (ASYNCcnt), bypassing VGPRs.
// Last K-step peeled so the steady-state loop is branch-free.
// ---------------------------------------------------------------------------
#define BM 128
#define BN 128
#define BK 32
#define LSTR 40   // bf16 elements per LDS row (32 + 8 pad)

// Issue one async tile-load group: 4 x b128 per thread (2 A chunks + 2 B chunks).
// GVS addressing: saddr = uniform base (SGPR pair), vaddr = per-lane byte offset,
// vdst = per-lane LDS byte address (low 32 bits of generic shared address).
__device__ __forceinline__ void issue_async_group(
    const bf16_t* __restrict__ A, const bf16_t* __restrict__ W,
    int m0, int n0, int K, int kbase,
    bf16_t* Asb, bf16_t* Bsb, int tid)
{
#pragma unroll
    for (int it = 0; it < 2; ++it) {
        const int chunk = tid + it * 256;
        const int row   = chunk >> 2;
        const int col   = (chunk & 3) * 8;
        const unsigned ldsA = (unsigned)(size_t)(Asb + row * LSTR + col);
        const unsigned offA =
            (unsigned)(((size_t)(m0 + row) * (size_t)K + (size_t)(kbase + col)) * 2u);
        asm volatile("global_load_async_to_lds_b128 %0, %1, %2"
                     :: "v"(ldsA), "v"(offA), "s"(A) : "memory");
        const unsigned ldsB = (unsigned)(size_t)(Bsb + row * LSTR + col);
        const unsigned offB =
            (unsigned)(((size_t)(n0 + row) * (size_t)K + (size_t)(kbase + col)) * 2u);
        asm volatile("global_load_async_to_lds_b128 %0, %1, %2"
                     :: "v"(ldsB), "v"(offB), "s"(W) : "memory");
    }
}

// One K-step of compute: 6 fragment loads from LDS + 8 WMMAs.
__device__ __forceinline__ void compute_step(
    const bf16_t* Asb, const bf16_t* Bsb,
    int waveM, int waveN, int half, int lr, v8f acc[2][4])
{
    FragBF af[2], bf[4];
    // A fragment: lane half=0 -> K {0..7,16..23}; half=1 -> K {8..15,24..31}
#pragma unroll
    for (int i = 0; i < 2; ++i) {
        const bf16_t* p = Asb + (waveM * 32 + i * 16 + lr) * LSTR;
        af[i].u[0] = *(const u32x4*)(p + half * 8);
        af[i].u[1] = *(const u32x4*)(p + 16 + half * 8);
    }
    // B fragment: lane half=0 -> K {0..7,16..23}; half=1 -> K {8..15,24..31}
#pragma unroll
    for (int j = 0; j < 4; ++j) {
        const bf16_t* p = Bsb + (waveN * 64 + j * 16 + lr) * LSTR;
        bf[j].u[0] = *(const u32x4*)(p + half * 8);
        bf[j].u[1] = *(const u32x4*)(p + 16 + half * 8);
    }
#pragma unroll
    for (int i = 0; i < 2; ++i)
#pragma unroll
        for (int j = 0; j < 4; ++j)
            acc[i][j] = __builtin_amdgcn_wmma_f32_16x16x32_bf16(
                false, af[i].v, false, bf[j].v,
                (short)0, acc[i][j], false, false);
}

__global__ void __launch_bounds__(256)
wmma_gemm_bf16(const bf16_t* __restrict__ A,    // [M,K]
               const bf16_t* __restrict__ W,    // [N,K]
               const float*  __restrict__ bias,   // [N] or nullptr
               const float*  __restrict__ resid,  // [M,N] or nullptr
               float*  __restrict__ C,            // [M,N]
               bf16_t* __restrict__ Cbf,          // optional bf16 copy of C
               int M, int N, int K, float alpha)
{
    __shared__ __align__(16) bf16_t As[2][BM * LSTR];
    __shared__ __align__(16) bf16_t Bs[2][BN * LSTR];

    const int tid   = threadIdx.x;
    const int wave  = tid >> 5;
    const int lane  = tid & 31;
    const int waveM = wave >> 1;    // 0..3
    const int waveN = wave & 1;     // 0..1
    const int half  = lane >> 4;    // 0/1
    const int lr    = lane & 15;

    const int m0 = blockIdx.y * BM;
    const int n0 = blockIdx.x * BN;

    v8f acc[2][4];
#pragma unroll
    for (int i = 0; i < 2; ++i)
#pragma unroll
        for (int j = 0; j < 4; ++j) acc[i][j] = (v8f)0.0f;

    const int nK = K / BK;

    // prologue: fill buffer 0
    issue_async_group(A, W, m0, n0, K, 0, As[0], Bs[0], tid);

    // steady state: branch-free body (last iteration peeled)
    for (int kb = 0; kb < nK - 1; ++kb) {
        const int buf = kb & 1;
        issue_async_group(A, W, m0, n0, K, (kb + 1) * BK,
                          As[buf ^ 1], Bs[buf ^ 1], tid);
        // 4 ops just issued stay outstanding; group kb retired (in-order)
        asm volatile("s_wait_asynccnt 0x4" ::: "memory");
        __syncthreads();   // all waves' tile data visible in LDS
        compute_step(As[buf], Bs[buf], waveM, waveN, half, lr, acc);
        __syncthreads();   // protect buf before it is overwritten next round
    }

    // epilogue K-step
    asm volatile("s_wait_asynccnt 0x0" ::: "memory");
    __syncthreads();
    compute_step(As[(nK - 1) & 1], Bs[(nK - 1) & 1], waveM, waveN, half, lr, acc);

    // Epilogue: C/D layout -> element r of v8f is (M = half*8 + r, N = lr)
#pragma unroll
    for (int i = 0; i < 2; ++i) {
#pragma unroll
        for (int j = 0; j < 4; ++j) {
            const int tm = m0 + waveM * 32 + i * 16 + half * 8;
            const int tn = n0 + waveN * 64 + j * 16 + lr;
            const float bv = bias ? bias[tn] : 0.0f;
#pragma unroll
            for (int r = 0; r < 8; ++r) {
                const size_t off = (size_t)(tm + r) * N + tn;
                float v = alpha * acc[i][j][r] + bv;
                if (resid) v += resid[off];
                C[off] = v;
                if (Cbf) Cbf[off] = (bf16_t)v;
            }
        }
    }
}

// ---------------------------------------------------------------------------
// LayerNorm (over H=4096) + exact GELU, f32 in -> bf16 out. One block per row.
// Single pass over global memory: 16 elements cached in registers per thread.
// ---------------------------------------------------------------------------
__global__ void __launch_bounds__(256)
ln_gelu_kernel(const float* __restrict__ h, const float* __restrict__ g,
               const float* __restrict__ be, bf16_t* __restrict__ out)
{
    const int H = 4096;
    __shared__ float red[256];
    const int row = blockIdx.x;
    const float* hr = h + (size_t)row * H;

    float loc[16];
    float s = 0.f;
#pragma unroll
    for (int j = 0; j < 16; ++j) {
        loc[j] = hr[threadIdx.x + j * 256];
        s += loc[j];
    }
    red[threadIdx.x] = s; __syncthreads();
    for (int off = 128; off > 0; off >>= 1) {
        if (threadIdx.x < off) red[threadIdx.x] += red[threadIdx.x + off];
        __syncthreads();
    }
    const float mu = red[0] * (1.0f / H);
    __syncthreads();

    float v = 0.f;
#pragma unroll
    for (int j = 0; j < 16; ++j) { const float d = loc[j] - mu; v += d * d; }
    red[threadIdx.x] = v; __syncthreads();
    for (int off = 128; off > 0; off >>= 1) {
        if (threadIdx.x < off) red[threadIdx.x] += red[threadIdx.x + off];
        __syncthreads();
    }
    const float rstd = rsqrtf(red[0] * (1.0f / H) + 1e-5f);

#pragma unroll
    for (int j = 0; j < 16; ++j) {
        const int i = threadIdx.x + j * 256;
        const float y  = (loc[j] - mu) * rstd * g[i] + be[i];
        const float ge = 0.5f * y * (1.0f + erff(y * 0.70710678118654752f));
        out[(size_t)row * H + i] = (bf16_t)ge;
    }
}

// ---------------------------------------------------------------------------
// Row softmax of L [S,S] -> bf16. One block per row.
// ---------------------------------------------------------------------------
__global__ void __launch_bounds__(256)
row_softmax_bf16(const float* __restrict__ L, bf16_t* __restrict__ out, int S)
{
    __shared__ float red[256];
    const int row = blockIdx.x;
    const float* lr = L + (size_t)row * S;

    float m = -INFINITY;
    for (int i = threadIdx.x; i < S; i += 256) m = fmaxf(m, lr[i]);
    red[threadIdx.x] = m; __syncthreads();
    for (int off = 128; off > 0; off >>= 1) {
        if (threadIdx.x < off) red[threadIdx.x] = fmaxf(red[threadIdx.x], red[threadIdx.x + off]);
        __syncthreads();
    }
    const float rmax = red[0]; __syncthreads();

    float s = 0.f;
    for (int i = threadIdx.x; i < S; i += 256) s += __expf(lr[i] - rmax);
    red[threadIdx.x] = s; __syncthreads();
    for (int off = 128; off > 0; off >>= 1) {
        if (threadIdx.x < off) red[threadIdx.x] += red[threadIdx.x + off];
        __syncthreads();
    }
    const float inv = 1.0f / red[0];

    for (int i = threadIdx.x; i < S; i += 256)
        out[(size_t)row * S + i] = (bf16_t)(__expf(lr[i] - rmax) * inv);
}

// ---------------------------------------------------------------------------
// softmax(inh/0.1) + top-60 (iterative LDS argmax) + straight-through output
// ---------------------------------------------------------------------------
#define SD 2048
#define KACT 60

__global__ void __launch_bounds__(256)
topk_sdr_kernel(const float* __restrict__ inh, float* __restrict__ out)
{
    __shared__ float vals[SD];
    __shared__ unsigned char hardf[SD];
    __shared__ float redv[256];
    __shared__ int   redi[256];

    const int row = blockIdx.x;
    const float* xr = inh + (size_t)row * SD;

    float lmax = -INFINITY;
    for (int i = threadIdx.x; i < SD; i += 256) {
        const float x = xr[i];
        vals[i] = x; hardf[i] = 0;
        lmax = fmaxf(lmax, x);
    }
    redv[threadIdx.x] = lmax; __syncthreads();
    for (int off = 128; off > 0; off >>= 1) {
        if (threadIdx.x < off) redv[threadIdx.x] = fmaxf(redv[threadIdx.x], redv[threadIdx.x + off]);
        __syncthreads();
    }
    const float rmax = redv[0]; __syncthreads();

    float lsum = 0.f;
    for (int i = threadIdx.x; i < SD; i += 256)
        lsum += __expf((vals[i] - rmax) * 10.0f);
    redv[threadIdx.x] = lsum; __syncthreads();
    for (int off = 128; off > 0; off >>= 1) {
        if (threadIdx.x < off) redv[threadIdx.x] += redv[threadIdx.x + off];
        __syncthreads();
    }
    const float rsum = redv[0]; __syncthreads();

    // iterative top-K argmax (ties -> lowest index, matching lax.top_k)
    for (int k = 0; k < KACT; ++k) {
        float bm = -INFINITY; int bi = SD;
        for (int i = threadIdx.x; i < SD; i += 256) {
            const float v = vals[i];
            if (v > bm) { bm = v; bi = i; }
        }
        redv[threadIdx.x] = bm; redi[threadIdx.x] = bi; __syncthreads();
        for (int off = 128; off > 0; off >>= 1) {
            if (threadIdx.x < off) {
                const float v2 = redv[threadIdx.x + off];
                const int   i2 = redi[threadIdx.x + off];
                if (v2 > redv[threadIdx.x] ||
                    (v2 == redv[threadIdx.x] && i2 < redi[threadIdx.x])) {
                    redv[threadIdx.x] = v2; redi[threadIdx.x] = i2;
                }
            }
            __syncthreads();
        }
        if (threadIdx.x == 0) { hardf[redi[0]] = 1; vals[redi[0]] = -INFINITY; }
        __syncthreads();
    }

    for (int i = threadIdx.x; i < SD; i += 256) {
        const float sm   = __expf((xr[i] - rmax) * 10.0f) / rsum;
        const float hard = hardf[i] ? 1.0f : 0.0f;
        out[(size_t)row * SD + i] = (hard - sm) + sm;
    }
}

// ---------------------------------------------------------------------------
// Launch
// ---------------------------------------------------------------------------
extern "C" void kernel_launch(void* const* d_in, const int* in_sizes, int n_in,
                              void* d_out, int out_size, void* d_ws, size_t ws_size,
                              hipStream_t stream) {
    const float* x   = (const float*)d_in[0];
    const float* W1  = (const float*)d_in[1];
    const float* b1  = (const float*)d_in[2];
    const float* g1  = (const float*)d_in[3];
    const float* be1 = (const float*)d_in[4];
    const float* W2  = (const float*)d_in[5];
    const float* b2  = (const float*)d_in[6];
    const float* L   = (const float*)d_in[7];

    const int B = 8192, D = 512, S = 2048, H = 4096;

    char* ws = (char*)d_ws;
    size_t off = 0;
    auto alloc = [&](size_t bytes) -> void* {
        void* p = ws + off;
        off += (bytes + 255) & ~(size_t)255;
        return p;
    };
    bf16_t* xb   = (bf16_t*)alloc((size_t)B * D * 2);
    bf16_t* w1b  = (bf16_t*)alloc((size_t)H * D * 2);
    bf16_t* w2b  = (bf16_t*)alloc((size_t)S * H * 2);
    bf16_t* lsb  = (bf16_t*)alloc((size_t)S * S * 2);
    bf16_t* hb   = (bf16_t*)alloc((size_t)B * H * 2);
    bf16_t* preb = (bf16_t*)alloc((size_t)B * S * 2);
    float*  h32  = (float*)alloc((size_t)B * H * 4);   // 128 MiB, reused below
    float*  pre32 = h32;                               // alias: h32 dead after LN/GELU
    float*  inh32 = (float*)((char*)h32 + (size_t)B * S * 4);

    // 1) bf16 conversions of GEMM operands
    {
        int n4;
        n4 = (B * D) / 4;
        cvt_f32_to_bf16<<<(n4 + 255) / 256, 256, 0, stream>>>(x, xb, n4);
        n4 = (H * D) / 4;
        cvt_f32_to_bf16<<<(n4 + 255) / 256, 256, 0, stream>>>(W1, w1b, n4);
        n4 = (S * H) / 4;
        cvt_f32_to_bf16<<<(n4 + 255) / 256, 256, 0, stream>>>(W2, w2b, n4);
    }

    // 2) GEMM1: h32 = x @ W1^T + b1     [B,H]
    wmma_gemm_bf16<<<dim3(H / BN, B / BM), 256, 0, stream>>>(
        xb, w1b, b1, nullptr, h32, nullptr, B, H, D, 1.0f);

    // 3) LayerNorm + GELU -> hb (bf16), single pass
    ln_gelu_kernel<<<B, 256, 0, stream>>>(h32, g1, be1, hb);

    // 4) GEMM2: pre32 = h @ W2^T + b2  [B,S]; also emit bf16 copy (fused cvt)
    wmma_gemm_bf16<<<dim3(S / BN, B / BM), 256, 0, stream>>>(
        hb, w2b, b2, nullptr, pre32, preb, B, S, H, 1.0f);

    // 5) Ls = softmax(L, rows) -> bf16
    row_softmax_bf16<<<S, 256, 0, stream>>>(L, lsb, S);

    // 6) GEMM3: inh32 = pre32 - 0.3 * (pre @ Ls^T)
    wmma_gemm_bf16<<<dim3(S / BN, B / BM), 256, 0, stream>>>(
        preb, lsb, nullptr, pre32, inh32, nullptr, B, S, S, -0.3f);

    // 7) softmax + top-60 + straight-through output
    topk_sdr_kernel<<<B, 256, 0, stream>>>(inh32, (float*)d_out);
}
